// FSAM_67473936220249
// MI455X (gfx1250) — matrix operand order repaired and checked by reading, not compiled, not run on hardware
//
#include <hip/hip_runtime.h>
#include <math.h>
#include <stdint.h>

// ---------------------------------------------------------------------------
// Types for gfx1250 WMMA (wave32): v16bf A/B fragments, v8f accumulator.
// ---------------------------------------------------------------------------
typedef __bf16 bf16;
typedef __attribute__((ext_vector_type(4)))  bf16  v4bf;
typedef __attribute__((ext_vector_type(8)))  bf16  v8bf;
typedef __attribute__((ext_vector_type(16))) bf16  v16bf;
typedef __attribute__((ext_vector_type(8)))  float v8f;

#define BDIM 8
#define SEQ  4096
#define CH   1024
#define NH   16
#define HD   64
#define HFF  2048
#define ROWS (BDIM * SEQ)   // 32768
#define LDK  48             // padded LDS K-stride (elements): 96B rows, 16B aligned

// --------------------------- CDNA5 async-to-LDS ----------------------------
// VDST = per-lane LDS byte address, VADDR = 64-bit global address.
__device__ __forceinline__ void cp_async_b128(unsigned lds_addr, const bf16* gsrc) {
  asm volatile("global_load_async_to_lds_b128 %0, %1, off"
               :
               : "v"(lds_addr), "v"(gsrc)
               : "memory");
}

#define WAIT_ASYNC(n) asm volatile("s_wait_asynccnt %0" ::"n"(n) : "memory")

__device__ __forceinline__ bf16 f2bf(float f) {
  unsigned u = __builtin_bit_cast(unsigned, f);
  u = (u + 0x7FFFu + ((u >> 16) & 1u)) >> 16;
  unsigned short h = (unsigned short)u;
  return __builtin_bit_cast(bf16, h);
}

__device__ __forceinline__ v8f wmma_bf16(v16bf a, v16bf b, v8f c) {
  return __builtin_amdgcn_wmma_f32_16x16x32_bf16(false, a, false, b,
                                                 (short)0, c, false, false);
}

// ---------------------------------------------------------------------------
// fp32 [K][N] -> bf16 transposed [N][K]  (weights; done once, tiny cost)
// ---------------------------------------------------------------------------
__global__ __launch_bounds__(256)
void f32_to_bf16_transpose(const float* __restrict__ a, bf16* __restrict__ o,
                           int K, int Nn) {
  __shared__ float tile[32][33];
  const int nbk = K >> 5;
  const int bk = blockIdx.x % nbk;
  const int bn = blockIdx.x / nbk;
  const int tx = threadIdx.x & 31, ty = threadIdx.x >> 5;  // ty: 0..7
#pragma unroll
  for (int i = 0; i < 32; i += 8)
    tile[ty + i][tx] = a[(size_t)(bk * 32 + ty + i) * Nn + bn * 32 + tx];
  __syncthreads();
#pragma unroll
  for (int i = 0; i < 32; i += 8)
    o[(size_t)(bn * 32 + ty + i) * K + bk * 32 + tx] = f2bf(tile[tx][ty + i]);
}

// ---------------------------------------------------------------------------
// LayerNorm over C=1024, one row per 256-thread block, bf16 output
// ---------------------------------------------------------------------------
__global__ __launch_bounds__(256)
void layernorm_to_bf16(const float* __restrict__ x, const float* __restrict__ g,
                       const float* __restrict__ b, bf16* __restrict__ y) {
  const int row = blockIdx.x;
  const int t = threadIdx.x;
  __shared__ float red[256];
  const float4 xv = ((const float4*)(x + (size_t)row * CH))[t];
  red[t] = xv.x + xv.y + xv.z + xv.w;
  __syncthreads();
  for (int o = 128; o > 0; o >>= 1) { if (t < o) red[t] += red[t + o]; __syncthreads(); }
  const float mu = red[0] * (1.0f / CH);
  __syncthreads();
  const float dx = xv.x - mu, dy = xv.y - mu, dz = xv.z - mu, dw = xv.w - mu;
  red[t] = dx * dx + dy * dy + dz * dz + dw * dw;
  __syncthreads();
  for (int o = 128; o > 0; o >>= 1) { if (t < o) red[t] += red[t + o]; __syncthreads(); }
  const float rs = rsqrtf(red[0] * (1.0f / CH) + 1e-5f);
  const float4 gv = ((const float4*)g)[t];
  const float4 bv = ((const float4*)b)[t];
  v4bf o4;
  o4[0] = f2bf(dx * rs * gv.x + bv.x);
  o4[1] = f2bf(dy * rs * gv.y + bv.y);
  o4[2] = f2bf(dz * rs * gv.z + bv.z);
  o4[3] = f2bf(dw * rs * gv.w + bv.w);
  *(v4bf*)(y + (size_t)row * CH + t * 4) = o4;
}

// ---------------------------------------------------------------------------
// bf16 WMMA GEMM: C = A(MxK) * BT(NxK)^T  [+bias][+res][gelu]
// 128x128 tile / block (8 waves, 2x4 WMMA tiles each), BK=32.
// Double-buffered via CDNA5 async global->LDS copies; pipeline unrolled by 2
// so buffer offsets are compile-time constants and the body is branch-free:
//   prefetch(other) -> s_wait_asynccnt 4 -> barrier -> 8x WMMA -> barrier
// K must be a multiple of 64 (holds: 1024 / 2048).
// out_mode: 0 = fp32 (Cf), 1 = bf16 (Cb), 2 = bf16 transposed (Cb[n*M+m])
// ---------------------------------------------------------------------------
__global__ __launch_bounds__(256)
void gemm_bf16_wmma(const bf16* __restrict__ A, const bf16* __restrict__ BT,
                    float* __restrict__ Cf, bf16* __restrict__ Cb,
                    const float* __restrict__ bias, const float* __restrict__ res,
                    int M, int K, int Nn, int do_gelu, int out_mode) {
  __shared__ bf16 As[2][128 * LDK];
  __shared__ bf16 Bs[2][128 * LDK];
  const int t = threadIdx.x;
  const int w = t >> 5, l = t & 31;
  const int lane16 = l & 15, hi = l >> 4;
  const int nbN = Nn >> 7;
  const int m0 = (blockIdx.x / nbN) << 7;
  const int n0 = (blockIdx.x % nbN) << 7;
  const int wr = w & 3, wc = w >> 2;

  // Staging: thread owns two 8-elem chunks of each tile (rows r0 and r0+64).
  const int r0 = t >> 2;          // 0..63
  const int kc = (t & 3) << 3;    // 0/8/16/24
  const bf16* pa0 = A + (size_t)(m0 + r0) * K + kc;
  const bf16* pa1 = pa0 + (size_t)64 * K;
  const bf16* pb0 = BT + (size_t)(n0 + r0) * K + kc;
  const bf16* pb1 = pb0 + (size_t)64 * K;
  const int o0 = r0 * LDK + kc;
  const int o1 = (r0 + 64) * LDK + kc;
  const unsigned lA0 = (unsigned)(uintptr_t)&As[0][o0];
  const unsigned lA1 = (unsigned)(uintptr_t)&As[0][o1];
  const unsigned lB0 = (unsigned)(uintptr_t)&Bs[0][o0];
  const unsigned lB1 = (unsigned)(uintptr_t)&Bs[0][o1];
  const unsigned BUFB = 128 * LDK * 2;  // byte distance buffer0 -> buffer1

  const v8f vzero = {0.f, 0.f, 0.f, 0.f, 0.f, 0.f, 0.f, 0.f};
  v8f acc[2][4];
#pragma unroll
  for (int i = 0; i < 2; ++i)
#pragma unroll
    for (int j = 0; j < 4; ++j) acc[i][j] = vzero;

  auto prefetch = [&](unsigned bofs) {
    cp_async_b128(lA0 + bofs, pa0);
    cp_async_b128(lA1 + bofs, pa1);
    cp_async_b128(lB0 + bofs, pb0);
    cp_async_b128(lB1 + bofs, pb1);
    pa0 += 32; pa1 += 32; pb0 += 32; pb1 += 32;
  };

  auto compute = [&](int buf) {
    v16bf af[2];
#pragma unroll
    for (int i = 0; i < 2; ++i) {
      const int m = wr * 32 + i * 16 + lane16;
      const int klo = hi * 16;
      v8bf lo = *(const v8bf*)&As[buf][m * LDK + klo];
      v8bf h8 = *(const v8bf*)&As[buf][m * LDK + 8 + klo];
#pragma unroll
      for (int e = 0; e < 8; ++e) { af[i][e] = lo[e]; af[i][e + 8] = h8[e]; }
    }
#pragma unroll
    for (int j = 0; j < 4; ++j) {
      const int n = wc * 64 + j * 16 + lane16;
      v16bf bv = *(const v16bf*)&Bs[buf][n * LDK + hi * 16];
#pragma unroll
      for (int i = 0; i < 2; ++i) acc[i][j] = wmma_bf16(af[i], bv, acc[i][j]);
    }
  };

  prefetch(0);
  for (int k0 = 0; k0 + 64 < K; k0 += 64) {
    prefetch(BUFB);
    WAIT_ASYNC(4);
    __syncthreads();
    compute(0);
    __syncthreads();
    prefetch(0);
    WAIT_ASYNC(4);
    __syncthreads();
    compute(1);
    __syncthreads();
  }
  // tail: buffers hold the last two K-steps (K % 64 == 0)
  prefetch(BUFB);
  WAIT_ASYNC(4);
  __syncthreads();
  compute(0);
  __syncthreads();
  WAIT_ASYNC(0);
  __syncthreads();
  compute(1);

  // epilogue: C/D layout -> lane = column n, VGPR r = row (hi lanes +8)
#pragma unroll
  for (int i = 0; i < 2; ++i)
#pragma unroll
    for (int j = 0; j < 4; ++j) {
      const int n = n0 + wc * 64 + j * 16 + lane16;
      const int mb = m0 + wr * 32 + i * 16 + hi * 8;
      const float bv = bias ? bias[n] : 0.0f;
      float vals[8];
#pragma unroll
      for (int r = 0; r < 8; ++r) {
        float val = acc[i][j][r] + bv;
        if (res) val += res[(size_t)(mb + r) * Nn + n];
        if (do_gelu) val = 0.5f * val * (1.0f + erff(val * 0.70710678118f));
        vals[r] = val;
      }
      if (out_mode == 0) {
#pragma unroll
        for (int r = 0; r < 8; ++r) Cf[(size_t)(mb + r) * Nn + n] = vals[r];
      } else if (out_mode == 1) {
#pragma unroll
        for (int r = 0; r < 8; ++r) Cb[(size_t)(mb + r) * Nn + n] = f2bf(vals[r]);
      } else {  // transposed bf16: lane-contiguous 16B store
        v8bf ov;
#pragma unroll
        for (int r = 0; r < 8; ++r) ov[r] = f2bf(vals[r]);
        *(v8bf*)(Cb + (size_t)n * M + mb) = ov;
      }
    }
}

// ---------------------------------------------------------------------------
// Attention scores: per (b,h), S(64x64) = (1/8) * Q^T K over N=4096.
// Inputs are pre-transposed qT/kT [C][B*N] -> staging is contiguous copies.
// Same branch-free double-buffered async pipeline (SEQ % 64 == 0).
// ---------------------------------------------------------------------------
__global__ __launch_bounds__(256)
void attn_scores_wmma(const bf16* __restrict__ qT, const bf16* __restrict__ kT,
                      float* __restrict__ S) {
  const int bh = blockIdx.x;
  const int b = bh >> 4, h = bh & 15;
  const int t = threadIdx.x;
  const int w = t >> 5, l = t & 31;
  const int lane16 = l & 15, hi = l >> 4;
  const int wr = w & 3, wc = w >> 2;  // 4x2 waves cover 64x64
  __shared__ bf16 Qs[2][64 * LDK];
  __shared__ bf16 Ks[2][64 * LDK];

  const int r0 = t >> 2;          // d-row 0..63
  const int kc = (t & 3) << 3;    // token-chunk 0/8/16/24
  const bf16* pq = qT + (size_t)(h * HD + r0) * ROWS + (size_t)b * SEQ + kc;
  const bf16* pk = kT + (size_t)(h * HD + r0) * ROWS + (size_t)b * SEQ + kc;
  const int oQ = r0 * LDK + kc;
  const unsigned lQ = (unsigned)(uintptr_t)&Qs[0][oQ];
  const unsigned lK = (unsigned)(uintptr_t)&Ks[0][oQ];
  const unsigned BUFB = 64 * LDK * 2;  // byte distance buffer0 -> buffer1

  const v8f vzero = {0.f, 0.f, 0.f, 0.f, 0.f, 0.f, 0.f, 0.f};
  v8f acc[2] = {vzero, vzero};

  auto prefetch = [&](unsigned bofs) {
    cp_async_b128(lQ + bofs, pq);
    cp_async_b128(lK + bofs, pk);
    pq += 32; pk += 32;
  };

  auto compute = [&](int buf) {
    v16bf af;
    const int m = wr * 16 + lane16;
    const int klo = hi * 16;
    v8bf lo = *(const v8bf*)&Qs[buf][m * LDK + klo];
    v8bf h8 = *(const v8bf*)&Qs[buf][m * LDK + 8 + klo];
#pragma unroll
    for (int e = 0; e < 8; ++e) { af[e] = lo[e]; af[e + 8] = h8[e]; }
#pragma unroll
    for (int j = 0; j < 2; ++j) {
      const int n = wc * 32 + j * 16 + lane16;
      v16bf bv = *(const v16bf*)&Ks[buf][n * LDK + hi * 16];
      acc[j] = wmma_bf16(af, bv, acc[j]);
    }
  };

  prefetch(0);
  for (int n0 = 0; n0 + 64 < SEQ; n0 += 64) {
    prefetch(BUFB);
    WAIT_ASYNC(2);
    __syncthreads();
    compute(0);
    __syncthreads();
    prefetch(0);
    WAIT_ASYNC(2);
    __syncthreads();
    compute(1);
    __syncthreads();
  }
  prefetch(BUFB);
  WAIT_ASYNC(2);
  __syncthreads();
  compute(0);
  __syncthreads();
  WAIT_ASYNC(0);
  __syncthreads();
  compute(1);

  float* Sb = S + (size_t)bh * (HD * HD);
#pragma unroll
  for (int j = 0; j < 2; ++j) {
    const int n = wc * 32 + j * 16 + lane16;
    const int mb = wr * 16 + hi * 8;
#pragma unroll
    for (int r = 0; r < 8; ++r) Sb[(mb + r) * HD + n] = acc[j][r] * 0.125f;
  }
}

// ---------------------------------------------------------------------------
// Softmax over 64-wide rows of S -> bf16 P
// ---------------------------------------------------------------------------
__global__ __launch_bounds__(64)
void attn_softmax(const float* __restrict__ S, bf16* __restrict__ P) {
  const int bh = blockIdx.x, r = threadIdx.x;
  const float* row = S + (size_t)bh * (HD * HD) + r * HD;
  float mx = -3.4e38f;
  for (int c = 0; c < HD; ++c) mx = fmaxf(mx, row[c]);
  float s = 0.f;
  for (int c = 0; c < HD; ++c) s += expf(row[c] - mx);
  const float inv = 1.0f / s;
  bf16* pr = P + (size_t)bh * (HD * HD) + r * HD;
  for (int c = 0; c < HD; ++c) pr[c] = f2bf(expf(row[c] - mx) * inv);
}

// ---------------------------------------------------------------------------
// O = P(64x64) * V^T(64xN) per (b,h); V fragments contiguous from global.
// Output bf16 into (B,N,C) layout for the proj GEMM.
// ---------------------------------------------------------------------------
__global__ __launch_bounds__(256)
void attn_out_wmma(const bf16* __restrict__ P, const bf16* __restrict__ v,
                   bf16* __restrict__ o) {
  const int nchunks = SEQ >> 8;  // 16
  const int bh = blockIdx.x / nchunks;
  const int nb = blockIdx.x % nchunks;
  const int b = bh >> 4, h = bh & 15;
  const int t = threadIdx.x;
  const int w = t >> 5, l = t & 31;
  const int lane16 = l & 15, hi = l >> 4;
  __shared__ bf16 Ps[64 * 72];  // [m][k], 144B row stride

  {
    const int m = t >> 2, kcp = (t & 3) << 4;  // 256 threads cover 64x64
    *(v8bf*)&Ps[m * 72 + kcp] = *(const v8bf*)(P + (size_t)bh * 4096 + m * 64 + kcp);
    *(v8bf*)&Ps[m * 72 + kcp + 8] =
        *(const v8bf*)(P + (size_t)bh * 4096 + m * 64 + kcp + 8);
  }
  __syncthreads();

  const v8f vzero = {0.f, 0.f, 0.f, 0.f, 0.f, 0.f, 0.f, 0.f};
  v8f acc[4][2];
#pragma unroll
  for (int i = 0; i < 4; ++i) { acc[i][0] = vzero; acc[i][1] = vzero; }

  const int n0 = (nb << 8) + w * 32;
  const size_t vbase = ((size_t)b * SEQ) * CH + h * HD;

#pragma unroll
  for (int ks = 0; ks < 2; ++ks) {
    const int kb = ks * 32;
    v16bf bv[2];
#pragma unroll
    for (int j = 0; j < 2; ++j) {
      const int n = n0 + j * 16 + lane16;
      bv[j] = *(const v16bf*)(v + vbase + (size_t)n * CH + kb + hi * 16);
    }
#pragma unroll
    for (int i = 0; i < 4; ++i) {
      const int m = i * 16 + lane16;
      const int klo = kb + hi * 16;
      v16bf af;
      v8bf lo = *(const v8bf*)&Ps[m * 72 + klo];
      v8bf h8 = *(const v8bf*)&Ps[m * 72 + klo + 8];
#pragma unroll
      for (int e = 0; e < 8; ++e) { af[e] = lo[e]; af[e + 8] = h8[e]; }
#pragma unroll
      for (int j = 0; j < 2; ++j) acc[i][j] = wmma_bf16(af, bv[j], acc[i][j]);
    }
  }

#pragma unroll
  for (int i = 0; i < 4; ++i)
#pragma unroll
    for (int j = 0; j < 2; ++j) {
      const int n = n0 + j * 16 + lane16;
      v8bf ov;
#pragma unroll
      for (int r = 0; r < 8; ++r) ov[r] = f2bf(acc[i][j][r]);
      *(v8bf*)(o + ((size_t)b * SEQ + n) * CH + h * HD + i * 16 + hi * 8) = ov;
    }
}

// ---------------------------------------------------------------------------
// Host launcher
// ---------------------------------------------------------------------------
extern "C" void kernel_launch(void* const* d_in, const int* in_sizes, int n_in,
                              void* d_out, int out_size, void* d_ws, size_t ws_size,
                              hipStream_t stream) {
  (void)in_sizes; (void)n_in; (void)out_size; (void)ws_size;
  const float* x      = (const float*)d_in[0];
  const float* ln1_g  = (const float*)d_in[1];
  const float* ln1_b  = (const float*)d_in[2];
  const float* ln2_g  = (const float*)d_in[3];
  const float* ln2_b  = (const float*)d_in[4];
  const float* wq     = (const float*)d_in[5];
  const float* wk     = (const float*)d_in[6];
  const float* wv     = (const float*)d_in[7];
  const float* proj_w = (const float*)d_in[8];
  const float* proj_b = (const float*)d_in[9];
  const float* fc1_w  = (const float*)d_in[10];
  const float* fc1_b  = (const float*)d_in[11];
  const float* fc2_w  = (const float*)d_in[12];
  const float* fc2_b  = (const float*)d_in[13];
  float* out = (float*)d_out;

  char* ws = (char*)d_ws;
  const size_t MB = 1ull << 20;
  bf16*  h_bf  = (bf16*)(ws + 0);          // 64MB: ln1 out, later ln2 out
  bf16*  qT    = (bf16*)(ws + 64 * MB);    // 64MB: q^T, later attention output o
  bf16*  kT    = (bf16*)(ws + 128 * MB);   // 64MB
  bf16*  v_bf  = (bf16*)(ws + 192 * MB);   // 64MB
  bf16*  o_bf  = qT;                       // reuse (q dead after scores)
  bf16*  g_bf  = kT;                       // 128MB spans k+v (dead after attn_out)
  float* x1    = (float*)(ws + 256 * MB);  // 128MB fp32 residual stream
  float* S     = (float*)(ws + 384 * MB);  // 2MB scores
  bf16*  P     = (bf16*)(ws + 387 * MB);   // 1MB probs
  bf16*  wqT   = (bf16*)(ws + 389 * MB);
  bf16*  wkT   = (bf16*)(ws + 391 * MB);
  bf16*  wvT   = (bf16*)(ws + 393 * MB);
  bf16*  pwT   = (bf16*)(ws + 395 * MB);
  bf16*  f1T   = (bf16*)(ws + 397 * MB);
  bf16*  f2T   = (bf16*)(ws + 401 * MB);

  // weights -> bf16, transposed to [N][K]
  f32_to_bf16_transpose<<<(CH / 32) * (CH / 32), 256, 0, stream>>>(wq, wqT, CH, CH);
  f32_to_bf16_transpose<<<(CH / 32) * (CH / 32), 256, 0, stream>>>(wk, wkT, CH, CH);
  f32_to_bf16_transpose<<<(CH / 32) * (CH / 32), 256, 0, stream>>>(wv, wvT, CH, CH);
  f32_to_bf16_transpose<<<(CH / 32) * (CH / 32), 256, 0, stream>>>(proj_w, pwT, CH, CH);
  f32_to_bf16_transpose<<<(CH / 32) * (HFF / 32), 256, 0, stream>>>(fc1_w, f1T, CH, HFF);
  f32_to_bf16_transpose<<<(HFF / 32) * (CH / 32), 256, 0, stream>>>(fc2_w, f2T, HFF, CH);

  layernorm_to_bf16<<<ROWS, 256, 0, stream>>>(x, ln1_g, ln1_b, h_bf);

  const int gQKV = (ROWS / 128) * (CH / 128);   // 2048
  // q,k written transposed [C][B*N] for the scores kernel; v normal
  gemm_bf16_wmma<<<gQKV, 256, 0, stream>>>(h_bf, wqT, nullptr, qT,
                                           nullptr, nullptr, ROWS, CH, CH, 0, 2);
  gemm_bf16_wmma<<<gQKV, 256, 0, stream>>>(h_bf, wkT, nullptr, kT,
                                           nullptr, nullptr, ROWS, CH, CH, 0, 2);
  gemm_bf16_wmma<<<gQKV, 256, 0, stream>>>(h_bf, wvT, nullptr, v_bf,
                                           nullptr, nullptr, ROWS, CH, CH, 0, 1);

  attn_scores_wmma<<<BDIM * NH, 256, 0, stream>>>(qT, kT, S);
  attn_softmax<<<BDIM * NH, 64, 0, stream>>>(S, P);
  attn_out_wmma<<<BDIM * NH * (SEQ / 256), 256, 0, stream>>>(P, v_bf, o_bf);

  gemm_bf16_wmma<<<gQKV, 256, 0, stream>>>(o_bf, pwT, x1, nullptr,
                                           proj_b, x, ROWS, CH, CH, 0, 0);

  layernorm_to_bf16<<<ROWS, 256, 0, stream>>>(x1, ln2_g, ln2_b, h_bf);

  const int gFC1 = (ROWS / 128) * (HFF / 128);  // 4096
  gemm_bf16_wmma<<<gFC1, 256, 0, stream>>>(h_bf, f1T, nullptr, g_bf,
                                           fc1_b, nullptr, ROWS, CH, HFF, 1, 1);
  gemm_bf16_wmma<<<gQKV, 256, 0, stream>>>(g_bf, f2T, out, nullptr,
                                           fc2_b, x1, ROWS, HFF, CH, 0, 0);
}
